// CommonsenseGRUModel_85186381349039
// MI455X (gfx1250) — matrix operand order repaired and checked by reading, not compiled, not run on hardware
//
#include <hip/hip_runtime.h>

// ---------------- problem constants (from reference) ----------------
#define T_DIM 100
#define B_DIM 64
#define P_DIM 2
#define D_M   1024
#define D_S   768
#define DG    150          // D_g = D_p = D_r = D_i
#define DE    100          // D_e
#define DH    100          // D_h
#define NCLS  7
#define TB    (T_DIM * B_DIM)   // 6400

typedef __attribute__((ext_vector_type(16))) _Float16 v16h;
typedef __attribute__((ext_vector_type(8)))  float    v8f;
typedef float    f4u   __attribute__((ext_vector_type(4),  aligned(4)));  // dword-aligned 4xf32
typedef _Float16 h8a4  __attribute__((ext_vector_type(8),  aligned(4)));  // dword-aligned 8xf16

enum { INIT_ZERO = 0, INIT_BIAS = 1, INIT_LOAD = 2 };

__device__ __forceinline__ float sigmoidf_(float x) { return 1.0f / (1.0f + __expf(-x)); }

// =====================================================================
// WMMA GEMM:  C[M,N] (+)= A[M,K](f32) * W[N,K](f16)^T  (+ bias / + Cin)
//   SPLIT=false: A is one row-major [M, astride] buffer (A0).
//   SPLIT=true : A is 4 chunk buffers of 1024 cols each (lin_in concat);
//                chunk selected once per iteration from uniform k0 (SALU).
//   RMAP=true  : rowmap[] gathers/zeroes A rows (implements _reverse_seq);
//                only this instantiation pays the zero-mask multiply.
//   W is PRE-CONVERTED f16, row-major [N, wld], cols woff..woff+K used.
//   B fragments load as raw f16 (2 x b128 per fragment, zero converts);
//   all 4 fragments are loaded before the 4 back-to-back WMMAs.
//   OOB B columns / A tail rows: loads clamped in-bounds, stores guarded.
//   One wave32 per 16(M)x64(N) tile; K-tail handled once after the loop.
// =====================================================================
template <bool SPLIT, bool RMAP>
__global__ __launch_bounds__(32)
void wmma_gemm_kernel(const float* __restrict__ A0, const float* __restrict__ A1,
                      const float* __restrict__ A2, const float* __restrict__ A3,
                      int astride,
                      const int* __restrict__ rowmap,
                      const _Float16* __restrict__ Wt, int wld, int woff,
                      const float* __restrict__ bias,
                      const float* __restrict__ Cin, int cinld,
                      float* __restrict__ C, int cld,
                      int M, int N, int K, int initmode, int act)
{
    const int lane = threadIdx.x & 31;
    const int hi   = lane >> 4;        // lane half (0/1)
    const int lo   = lane & 15;
    const int m0   = blockIdx.y << 4;  // 16-row tile
    const int n0   = blockIdx.x << 6;  // 64-col tile (4 x 16)

    // ---- resolve this lane's A row (A-fragment rows ride on lanes lo)
    const int am = m0 + lo;
    long srow = (am < M) ? am : 0;     // clamp tail rows (store-guarded anyway)
    float amul = 1.0f;
    if (RMAP) {
        int rm = (am < M) ? rowmap[am] : -1;
        amul = (rm >= 0) ? 1.0f : 0.0f;   // -1 rows must contribute exactly 0
        srow = (rm >= 0) ? rm : 0;
    }
    const long arow = srow * (long)astride;

    // ---- clamped B row pointers (OOB columns are store-guarded, no mask)
    const _Float16* bp[4];
#pragma unroll
    for (int g = 0; g < 4; ++g) {
        const int n = n0 + (g << 4) + lo;
        const int nc = (n < N) ? n : (N - 1);
        bp[g] = Wt + (long)nc * wld + woff;
    }

    // ---- accumulator init (ISA C/D layout: lane lo -> col, vgpr i -> row hi*8+i)
    v8f acc[4];
#pragma unroll
    for (int g = 0; g < 4; ++g) {
        const int n = n0 + (g << 4) + lo;
#pragma unroll
        for (int i = 0; i < 8; ++i) {
            float v = 0.0f;
            if (initmode == INIT_BIAS) {
                if (n < N) v = bias[n];
            } else if (initmode == INIT_LOAD) {
                const int mm = m0 + (hi << 3) + i;
                if (n < N && mm < M) v = Cin[(long)mm * cinld + n];
            }
            acc[g][i] = v;
        }
    }

    // ---- branch-free main loop over full 32-wide K tiles
    const int Kfull = K & ~31;
    for (int k0 = 0; k0 < Kfull; k0 += 32) {
        const float* abase;
        if (SPLIT) {
            const int sel = k0 >> 10;   // uniform -> scalar select, no exec masks
            const float* ab = (sel == 0) ? A0 : (sel == 1) ? A1 : (sel == 2) ? A2 : A3;
            abase = ab + arow + (k0 & 1023);
        } else {
            abase = A0 + arow + k0;
        }
        // A raw loads (16x32 f16 fragment; lane half 'hi' holds runs hi*8 and 16+hi*8)
        const float* ap = abase + (hi << 3);
        f4u a0 = *(const f4u*)ap;
        f4u a1 = *(const f4u*)(ap + 4);
        f4u a2 = *(const f4u*)(ap + 16);
        f4u a3 = *(const f4u*)(ap + 20);
        // B fragments: runs of 8 f16 per lane at row n, k0 + hi*8 and k0 + 16 + hi*8 (2 x b128 each)
        v16h bfr[4];
#pragma unroll
        for (int g = 0; g < 4; ++g) {
            const _Float16* p = bp[g] + k0 + (hi << 3);
            h8a4 u0 = *(const h8a4*)p;
            h8a4 u1 = *(const h8a4*)(p + 16);
#pragma unroll
            for (int j = 0; j < 8; ++j) { bfr[g][j] = u0[j]; bfr[g][8 + j] = u1[j]; }
            __builtin_prefetch(p + 32, 0, 3);   // next K tile of this W row
        }
        // convert A to f16 (packed converts)
        v16h af;
#pragma unroll
        for (int j = 0; j < 4; ++j) {
            if (RMAP) {
                af[j]      = (_Float16)(a0[j] * amul);
                af[4 + j]  = (_Float16)(a1[j] * amul);
                af[8 + j]  = (_Float16)(a2[j] * amul);
                af[12 + j] = (_Float16)(a3[j] * amul);
            } else {
                af[j]      = (_Float16)a0[j];
                af[4 + j]  = (_Float16)a1[j];
                af[8 + j]  = (_Float16)a2[j];
                af[12 + j] = (_Float16)a3[j];
            }
        }
        // 4 back-to-back WMMAs (independent accumulators, no D-overlap hazards)
#pragma unroll
        for (int g = 0; g < 4; ++g) {
            acc[g] = __builtin_amdgcn_wmma_f32_16x16x32_f16(
                false, af, false, bfr[g], (short)0, acc[g], false, false);
        }
    }

    // ---- single guarded tail tile (K % 32), executed at most once
    if (Kfull < K) {
        const int k0 = Kfull;
        v16h af;
#pragma unroll
        for (int h = 0; h < 2; ++h) {
#pragma unroll
            for (int j = 0; j < 8; ++j) {
                const int k = k0 + (hi << 3) + (h << 4) + j;
                float v = 0.0f;
                if (k < K) {
                    if (SPLIT) {
                        const int sel = k >> 10;
                        const float* ab = (sel == 0) ? A0 : (sel == 1) ? A1 : (sel == 2) ? A2 : A3;
                        v = ab[arow + (k & 1023)];
                    } else {
                        v = A0[arow + k];
                    }
                }
                af[h * 8 + j] = (_Float16)(RMAP ? (v * amul) : v);
            }
        }
#pragma unroll
        for (int g = 0; g < 4; ++g) {
            v16h bf;
#pragma unroll
            for (int j = 0; j < 16; ++j) {
                const int k = k0 + (hi << 3) + ((j >> 3) << 4) + (j & 7);
                bf[j] = (k < K) ? bp[g][k] : (_Float16)0.0f;
            }
            acc[g] = __builtin_amdgcn_wmma_f32_16x16x32_f16(
                false, af, false, bf, (short)0, acc[g], false, false);
        }
    }

    // ---- epilogue
#pragma unroll
    for (int g = 0; g < 4; ++g) {
        const int n = n0 + (g << 4) + lo;
        if (n < N) {
#pragma unroll
            for (int i = 0; i < 8; ++i) {
                const int mm = m0 + (hi << 3) + i;
                if (mm < M) {
                    float v = acc[g][i];
                    if (act) v = fmaxf(v, 0.0f);
                    C[(long)mm * cld + n] = v;
                }
            }
        }
    }
}

// =====================================================================
// Elementwise / scan helper kernels
// =====================================================================
__global__ void zero_kernel(float* __restrict__ p, int n)
{
    int i = blockIdx.x * blockDim.x + threadIdx.x;
    if (i < n) p[i] = 0.0f;
}

// one-time f32 -> f16 weight conversion
__global__ void f32to16_kernel(const float* __restrict__ s, _Float16* __restrict__ d, int n)
{
    int i = blockIdx.x * blockDim.x + threadIdx.x;
    if (i < n) d[i] = (_Float16)s[i];
}

// rowmap[t*B+b] = flat row of the reversed sequence element, or -1 if invalid
__global__ void rowmap_kernel(const float* __restrict__ umask, int* __restrict__ rowmap)
{
    int idx = blockIdx.x * blockDim.x + threadIdx.x;
    if (idx >= TB) return;
    int t = idx / B_DIM, b = idx - t * B_DIM;
    float c = 0.0f;
    for (int tt = 0; tt < T_DIM; ++tt) c += umask[b * T_DIM + tt];
    int ci = (int)(c + 0.5f);
    int id = ci - 1 - t;
    rowmap[idx] = (id >= 0 && id < T_DIM) ? (id * B_DIM + b) : -1;
}

// qidx[b] = argmax(qmask[t,b,:]); dst[b,0:150]=q0[b,qidx], dst[b,150:300]=r0[b,qidx]
__global__ void prep_g_kernel(const float* __restrict__ q0, const float* __restrict__ r0,
                              const float* __restrict__ qmask, const int* __restrict__ rowmap,
                              int t, int* __restrict__ qidx, float* __restrict__ dst)
{
    const int b = blockIdx.x;
    __shared__ int qs;
    if (threadIdx.x == 0) {
        int row = t * B_DIM + b;
        if (rowmap) row = rowmap[row];
        float a = 0.0f, c = 0.0f;
        if (row >= 0) { a = qmask[row * P_DIM]; c = qmask[row * P_DIM + 1]; }
        int qi = (c > a) ? 1 : 0;
        qs = qi;
        qidx[b] = qi;
    }
    __syncthreads();
    const int qi = qs;
    for (int d = threadIdx.x; d < DG; d += blockDim.x) {
        dst[b * 300 + d]      = q0[(b * P_DIM + qi) * DG + d];
        dst[b * 300 + DG + d] = r0[(b * P_DIM + qi) * DG + d];
    }
}

__global__ void prep_selq_kernel(const float* __restrict__ q0, const int* __restrict__ qidx,
                                 float* __restrict__ dst)
{
    const int b = blockIdx.x;
    const int qi = qidx[b];
    for (int d = threadIdx.x; d < DG; d += blockDim.x)
        dst[b * DG + d] = q0[(b * P_DIM + qi) * DG + d];
}

__global__ void prep_e_kernel(const float* __restrict__ q0, const float* __restrict__ r0,
                              const float* __restrict__ i0, const int* __restrict__ qidx,
                              float* __restrict__ dst)
{
    const int b = blockIdx.x;
    const int qi = qidx[b];
    for (int d = threadIdx.x; d < DG; d += blockDim.x) {
        dst[b * 450 + d]          = q0[(b * P_DIM + qi) * DG + d];
        dst[b * 450 + DG + d]     = r0[(b * P_DIM + qi) * DG + d];
        dst[b * 450 + 2 * DG + d] = i0[(b * P_DIM + qi) * DG + d];
    }
}

// plain GRU gate combine: hout = (1-z)*n + z*hprev
__global__ void gru_combine_kernel(const float* __restrict__ gi, const float* __restrict__ gh,
                                   const float* __restrict__ hprev, float* __restrict__ hout,
                                   int rows, int dh)
{
    int idx = blockIdx.x * blockDim.x + threadIdx.x;
    if (idx >= rows * dh) return;
    int r = idx / dh, d = idx - r * dh;
    const float* gir = gi + (long)r * 3 * dh;
    const float* ghr = gh + (long)r * 3 * dh;
    float rg = sigmoidf_(gir[d] + ghr[d]);
    float zg = sigmoidf_(gir[dh + d] + ghr[dh + d]);
    float ng = tanhf(gir[2 * dh + d] + rg * ghr[2 * dh + d]);
    hout[idx] = (1.0f - zg) * ng + zg * hprev[idx];
}

// per-party GRU combine + qmask-masked state update (in place on h[B,P,dh]);
// gi shared across parties (rows [B,3dh]), gh per (b,p) (rows [B*P,3dh])
__global__ void gru_combine_party_kernel(const float* __restrict__ gi, const float* __restrict__ gh,
                                         float* __restrict__ h, const float* __restrict__ qmask,
                                         const int* __restrict__ rowmap, int t, int dh)
{
    int idx = blockIdx.x * blockDim.x + threadIdx.x;
    if (idx >= B_DIM * P_DIM * dh) return;
    int d = idx % dh;
    int bp = idx / dh;
    int b = bp >> 1, p = bp & 1;
    const float* gir = gi + (long)b * 3 * dh;
    const float* ghr = gh + (long)bp * 3 * dh;
    float rg = sigmoidf_(gir[d] + ghr[d]);
    float zg = sigmoidf_(gir[dh + d] + ghr[dh + d]);
    float ng = tanhf(gir[2 * dh + d] + rg * ghr[2 * dh + d]);
    float hp = h[idx];
    float hn = (1.0f - zg) * ng + zg * hp;
    int row = t * B_DIM + b;
    if (rowmap) row = rowmap[row];
    float qm = (row >= 0) ? qmask[row * P_DIM + p] : 0.0f;
    h[idx] = hp * (1.0f - qm) + hn * qm;
}

// e-cell combine: updates e0 in place and writes e_out[t]
__global__ void gru_combine_e_kernel(const float* __restrict__ gi, const float* __restrict__ gh,
                                     float* __restrict__ e0, float* __restrict__ eout, int dh)
{
    int idx = blockIdx.x * blockDim.x + threadIdx.x;
    if (idx >= B_DIM * dh) return;
    int r = idx / dh, d = idx - r * dh;
    const float* gir = gi + (long)r * 3 * dh;
    const float* ghr = gh + (long)r * 3 * dh;
    float rg = sigmoidf_(gir[d] + ghr[d]);
    float zg = sigmoidf_(gir[dh + d] + ghr[dh + d]);
    float ng = tanhf(gir[2 * dh + d] + rg * ghr[2 * dh + d]);
    float hn = (1.0f - zg) * ng + zg * e0[idx];
    e0[idx] = hn;
    eout[idx] = hn;
}

// attention over g_hist rows t' < t: cout[b,:] = sum_t' softmax(g_hist.att_w)[t'] * g_hist[t',b,:]
// att_w is staged into LDS via CDNA5 async global->LDS copy (ASYNCcnt).
__global__ void attention_kernel(const float* __restrict__ ghist, const float* __restrict__ attw,
                                 float* __restrict__ cout, int t)
{
    __shared__ float sw[DG];
    __shared__ float sc[T_DIM];
    __shared__ float red[128];
    const int b = blockIdx.x;
    const int tid = threadIdx.x;

#if defined(__gfx1250__)
    for (int d = tid; d < DG; d += blockDim.x) {
        unsigned ldsoff = (unsigned)(unsigned long long)(&sw[d]);   // LDS byte offset
        const float* gp = attw + d;
        asm volatile("global_load_async_to_lds_b32 %0, %1, off"
                     :: "v"(ldsoff), "v"(gp) : "memory");
    }
    asm volatile("s_wait_asynccnt 0x0" ::: "memory");
#else
    for (int d = tid; d < DG; d += blockDim.x) sw[d] = attw[d];
#endif
    __syncthreads();

    if (t == 0) {
        for (int d = tid; d < DG; d += blockDim.x) cout[b * DG + d] = 0.0f;
        return;
    }
    for (int tp = tid; tp < t; tp += blockDim.x) {
        const float* g = ghist + ((long)tp * B_DIM + b) * DG;
        float s = 0.0f;
        for (int d = 0; d < DG; ++d) s += g[d] * sw[d];
        sc[tp] = s;
    }
    __syncthreads();
    float mx = -3.0e38f;
    for (int tp = tid; tp < t; tp += blockDim.x) mx = fmaxf(mx, sc[tp]);
    red[tid] = mx;
    __syncthreads();
    for (int s = 64; s > 0; s >>= 1) { if (tid < s) red[tid] = fmaxf(red[tid], red[tid + s]); __syncthreads(); }
    mx = red[0];
    __syncthreads();
    float sm = 0.0f;
    for (int tp = tid; tp < t; tp += blockDim.x) { float e = __expf(sc[tp] - mx); sc[tp] = e; sm += e; }
    red[tid] = sm;
    __syncthreads();
    for (int s = 64; s > 0; s >>= 1) { if (tid < s) red[tid] += red[tid + s]; __syncthreads(); }
    const float inv = 1.0f / red[0];
    __syncthreads();
    for (int d = tid; d < DG; d += blockDim.x) {
        float a = 0.0f;
        for (int tp = 0; tp < t; ++tp) a += sc[tp] * ghist[((long)tp * B_DIM + b) * DG + d];
        cout[b * DG + d] = a * inv;
    }
}

// emotions[m, 0:100] = ef[m]; emotions[m, 100:200] = reversed(eb_scan)[m]
__global__ void emotions_kernel(const float* __restrict__ ef, const float* __restrict__ ebs,
                                const int* __restrict__ rowmap, float* __restrict__ em)
{
    int idx = blockIdx.x * blockDim.x + threadIdx.x;
    if (idx >= TB * DE) return;
    int m = idx / DE, d = idx - m * DE;
    em[(long)m * (2 * DE) + d] = ef[idx];
    int row = rowmap[m];
    em[(long)m * (2 * DE) + DE + d] = (row >= 0) ? ebs[(long)row * DE + d] : 0.0f;
}

// fused fc (7x100) + log_softmax
__global__ void fc_logsoftmax_kernel(const float* __restrict__ hidden, const float* __restrict__ fcw,
                                     const float* __restrict__ fcb, float* __restrict__ out)
{
    int m = blockIdx.x * blockDim.x + threadIdx.x;
    if (m >= TB) return;
    float lg[NCLS];
    float mx = -3.0e38f;
    const float* hr = hidden + (long)m * DH;
    for (int c = 0; c < NCLS; ++c) {
        float s = fcb[c];
        const float* wr = fcw + c * DH;
        for (int k = 0; k < DH; ++k) s += wr[k] * hr[k];
        lg[c] = s;
        mx = fmaxf(mx, s);
    }
    float sm = 0.0f;
    for (int c = 0; c < NCLS; ++c) sm += __expf(lg[c] - mx);
    float lse = mx + __logf(sm);
    for (int c = 0; c < NCLS; ++c) out[(long)m * NCLS + c] = lg[c] - lse;
}

// =====================================================================
// Host orchestration
// =====================================================================
struct GruH { const _Float16 *wih, *whh; const float *bih, *bhh; };
struct CsH  { GruH g, p, r, i, e; const float* att; };

extern "C" void kernel_launch(void* const* d_in, const int* in_sizes, int n_in,
                              void* d_out, int out_size, void* d_ws, size_t ws_size,
                              hipStream_t stream)
{
    (void)in_sizes; (void)n_in; (void)out_size; (void)ws_size;

    // ---- inputs (setup_inputs insertion order; params pytree in insertion order)
    const float* r1 = (const float*)d_in[0];
    const float* r2 = (const float*)d_in[1];
    const float* r3 = (const float*)d_in[2];
    const float* r4 = (const float*)d_in[3];
    const float* x1 = (const float*)d_in[4];
    const float* x2 = (const float*)d_in[5];
    const float* x3 = (const float*)d_in[6];
    // d_in[7], d_in[8] = o1, o2 (unused by reference)
    const float* qmaskp = (const float*)d_in[9];
    const float* umaskp = (const float*)d_in[10];
    const float* lin_in_w = (const float*)d_in[53];
    const float* lin_in_b = (const float*)d_in[54];
    const float* lin_w    = (const float*)d_in[55];
    const float* lin_b    = (const float*)d_in[56];
    const float* fc_w     = (const float*)d_in[57];
    const float* fc_b     = (const float*)d_in[58];

    // ---- workspace carving
    float* wsf = (float*)d_ws;
    size_t off = 0;
    auto alloc = [&](size_t n) { float* p = wsf + off; off += n; return p; };
    float* U     = alloc((size_t)TB * DH);
    float* pG    = alloc((size_t)TB * 450);
    float* pP    = alloc((size_t)TB * 450);
    float* pR    = alloc((size_t)TB * 450);
    float* pI    = alloc((size_t)TB * 450);
    float* pE    = alloc((size_t)TB * 300);
    float* ef    = alloc((size_t)TB * DE);
    float* ebs   = alloc((size_t)TB * DE);
    float* ghist = alloc((size_t)TB * DG);
    float* q0    = alloc((size_t)B_DIM * P_DIM * DG);
    float* r0    = alloc((size_t)B_DIM * P_DIM * DG);
    float* i0    = alloc((size_t)B_DIM * P_DIM * DG);
    float* e0    = alloc((size_t)B_DIM * DE);
    float* ghG   = alloc((size_t)B_DIM * 450);
    float* ghP   = alloc((size_t)B_DIM * P_DIM * 450);
    float* ghR   = alloc((size_t)B_DIM * P_DIM * 450);
    float* ghI   = alloc((size_t)B_DIM * P_DIM * 450);
    float* ghE   = alloc((size_t)B_DIM * 300);
    float* cbuf  = alloc((size_t)B_DIM * DG);
    float* gin   = alloc((size_t)B_DIM * 300);
    float* selq  = alloc((size_t)B_DIM * DG);
    float* erec  = alloc((size_t)B_DIM * 450);
    float* emo   = alloc((size_t)TB * 2 * DE);
    float* hid   = alloc((size_t)TB * DH);
    int* rowmapb = (int*)(wsf + off); off += TB;
    int* qidx    = (int*)(wsf + off); off += B_DIM;

    // f16 weight staging area + one-time conversion
    auto cvt16 = [&](const float* s, int n) {
        _Float16* d = (_Float16*)(wsf + off);
        off += (size_t)(n + 1) / 2;
        f32to16_kernel<<<(n + 255) / 256, 256, 0, stream>>>(s, d, n);
        return (const _Float16*)d;
    };
    auto gru16 = [&](int i, int kih) {   // kih = wih column count
        GruH g;
        g.wih = cvt16((const float*)d_in[i],     (i >= 32 ? 1 : 1) * 0 + 3 * ((i - 11) % 4 == 0 ? 0 : 0) + kih);  // placeholder, replaced below
        return g;
    };
    (void)gru16;
    auto make_gru = [&](int i, int rows, int kih, int dh) {
        GruH g;
        g.wih = cvt16((const float*)d_in[i],     rows * kih);
        g.whh = cvt16((const float*)d_in[i + 1], rows * dh);
        g.bih = (const float*)d_in[i + 2];
        g.bhh = (const float*)d_in[i + 3];
        return g;
    };
    auto make_cs = [&](int base) {
        CsH c;
        c.g = make_gru(base + 0,  450, 400,  DG);
        c.p = make_gru(base + 4,  450, 918,  DG);
        c.r = make_gru(base + 8,  450, 1018, DG);
        c.i = make_gru(base + 12, 450, 918,  DG);
        c.e = make_gru(base + 16, 300, 550,  DE);
        c.att = (const float*)d_in[base + 20];
        return c;
    };
    CsH fpar = make_cs(11);
    CsH bpar = make_cs(32);
    const _Float16* lin_in_wh = cvt16(lin_in_w, DH * 4 * D_M);
    const _Float16* lin_wh    = cvt16(lin_w,    DH * 2 * DE);

    auto zero = [&](float* p, int n) {
        zero_kernel<<<(n + 255) / 256, 256, 0, stream>>>(p, n);
    };
    // generic single-source GEMM launch; picks the RMAP instantiation only when needed
    auto gemm = [&](const float* A, int astride, const int* rmap,
                    const _Float16* W, int wld, int woff, const float* bias,
                    const float* Cin, int cinld, float* C, int cld,
                    int M, int N, int K, int init, int act) {
        dim3 grid((N + 63) / 64, (M + 15) / 16);
        if (rmap) {
            wmma_gemm_kernel<false, true><<<grid, 32, 0, stream>>>(
                A, A, A, A, astride, rmap, W, wld, woff, bias, Cin, cinld,
                C, cld, M, N, K, init, act);
        } else {
            wmma_gemm_kernel<false, false><<<grid, 32, 0, stream>>>(
                A, A, A, A, astride, nullptr, W, wld, woff, bias, Cin, cinld,
                C, cld, M, N, K, init, act);
        }
    };

    // ---- U = concat(r1..r4) @ lin_in_w^T + b   (A split in 1024-wide chunks)
    {
        dim3 grid((DH + 63) / 64, (TB + 15) / 16);
        wmma_gemm_kernel<true, false><<<grid, 32, 0, stream>>>(
            r1, r2, r3, r4, D_M, nullptr, lin_in_wh, 4 * D_M, 0, lin_in_b,
            nullptr, 0, U, DH, TB, DH, 4 * D_M, INIT_BIAS, 0);
    }

    // ---- one direction of the CommonsenseRNN scan
    auto run_dir = [&](const CsH& cp, const int* rmap, float* eout) {
        // time-parallel input projections (hoisted out of the scan), bias folded in
        gemm(U,  DH,  rmap, cp.g.wih, 400,  0,  cp.g.bih, nullptr, 0, pG, 450, TB, 450, DH,  INIT_BIAS, 0);
        gemm(U,  DH,  rmap, cp.r.wih, 1018, 0,  cp.r.bih, nullptr, 0, pR, 450, TB, 450, DH,  INIT_BIAS, 0);
        gemm(x2, D_S, rmap, cp.r.wih, 1018, DH, nullptr,  pR, 450,     pR, 450, TB, 450, D_S, INIT_LOAD, 0);
        gemm(x1, D_S, rmap, cp.p.wih, 918,  0,  cp.p.bih, nullptr, 0, pP, 450, TB, 450, D_S, INIT_BIAS, 0);
        gemm(x3, D_S, rmap, cp.i.wih, 918,  0,  cp.i.bih, nullptr, 0, pI, 450, TB, 450, D_S, INIT_BIAS, 0);
        gemm(U,  DH,  rmap, cp.e.wih, 550,  0,  cp.e.bih, nullptr, 0, pE, 300, TB, 300, DH,  INIT_BIAS, 0);

        zero(ghist, TB * DG);
        zero(q0, B_DIM * P_DIM * DG);
        zero(r0, B_DIM * P_DIM * DG);
        zero(i0, B_DIM * P_DIM * DG);
        zero(e0, B_DIM * DE);

        for (int t = 0; t < T_DIM; ++t) {
            float* giG = pG + (size_t)t * B_DIM * 450;
            float* giP = pP + (size_t)t * B_DIM * 450;
            float* giR = pR + (size_t)t * B_DIM * 450;
            float* giI = pI + (size_t)t * B_DIM * 450;
            float* giE = pE + (size_t)t * B_DIM * 300;
            const float* gprev = ghist + (size_t)((t > 0) ? (t - 1) : 0) * B_DIM * DG;
            float* gcur = ghist + (size_t)t * B_DIM * DG;

            // ---- g cell
            prep_g_kernel<<<B_DIM, 128, 0, stream>>>(q0, r0, qmaskp, rmap, t, qidx, gin);
            gemm(gin, 300, nullptr, cp.g.wih, 400, 100, nullptr, giG, 450, giG, 450,
                 B_DIM, 450, 300, INIT_LOAD, 0);
            gemm(gprev, DG, nullptr, cp.g.whh, DG, 0, cp.g.bhh, nullptr, 0, ghG, 450,
                 B_DIM, 450, DG, INIT_BIAS, 0);
            gru_combine_kernel<<<(B_DIM * DG + 255) / 256, 256, 0, stream>>>(
                giG, ghG, gprev, gcur, B_DIM, DG);

            // ---- attention context (only rows t' < t contribute)
            attention_kernel<<<B_DIM, 128, 0, stream>>>(ghist, cp.att, cbuf, t);

            // ---- r and p cells (gi shared across parties, gh per party)
            gemm(cbuf, DG, nullptr, cp.r.wih, 1018, 868, nullptr, giR, 450, giR, 450,
                 B_DIM, 450, DG, INIT_LOAD, 0);
            gemm(cbuf, DG, nullptr, cp.p.wih, 918, 768, nullptr, giP, 450, giP, 450,
                 B_DIM, 450, DG, INIT_LOAD, 0);
            gemm(r0, DG, nullptr, cp.r.whh, DG, 0, cp.r.bhh, nullptr, 0, ghR, 450,
                 B_DIM * P_DIM, 450, DG, INIT_BIAS, 0);
            gemm(q0, DG, nullptr, cp.p.whh, DG, 0, cp.p.bhh, nullptr, 0, ghP, 450,
                 B_DIM * P_DIM, 450, DG, INIT_BIAS, 0);
            gru_combine_party_kernel<<<(B_DIM * P_DIM * DG + 255) / 256, 256, 0, stream>>>(
                giR, ghR, r0, qmaskp, rmap, t, DG);
            gru_combine_party_kernel<<<(B_DIM * P_DIM * DG + 255) / 256, 256, 0, stream>>>(
                giP, ghP, q0, qmaskp, rmap, t, DG);

            // ---- i cell (uses updated q)
            prep_selq_kernel<<<B_DIM, 128, 0, stream>>>(q0, qidx, selq);
            gemm(selq, DG, nullptr, cp.i.wih, 918, 768, nullptr, giI, 450, giI, 450,
                 B_DIM, 450, DG, INIT_LOAD, 0);
            gemm(i0, DG, nullptr, cp.i.whh, DG, 0, cp.i.bhh, nullptr, 0, ghI, 450,
                 B_DIM * P_DIM, 450, DG, INIT_BIAS, 0);
            gru_combine_party_kernel<<<(B_DIM * P_DIM * DG + 255) / 256, 256, 0, stream>>>(
                giI, ghI, i0, qmaskp, rmap, t, DG);

            // ---- e cell (uses updated q, r, i)
            prep_e_kernel<<<B_DIM, 128, 0, stream>>>(q0, r0, i0, qidx, erec);
            gemm(erec, 450, nullptr, cp.e.wih, 550, 100, nullptr, giE, 300, giE, 300,
                 B_DIM, 300, 450, INIT_LOAD, 0);
            gemm(e0, DE, nullptr, cp.e.whh, DE, 0, cp.e.bhh, nullptr, 0, ghE, 300,
                 B_DIM, 300, DE, INIT_BIAS, 0);
            gru_combine_e_kernel<<<(B_DIM * DE + 255) / 256, 256, 0, stream>>>(
                giE, ghE, e0, eout + (size_t)t * B_DIM * DE, DE);
        }
    };

    // forward direction
    run_dir(fpar, nullptr, ef);
    // backward direction: rowmap implements _reverse_seq inside the GEMM A-loads
    rowmap_kernel<<<(TB + 255) / 256, 256, 0, stream>>>(umaskp, rowmapb);
    run_dir(bpar, rowmapb, ebs);

    // ---- head: concat(ef, rev(eb_scan)) -> linear+relu -> fc -> log_softmax
    emotions_kernel<<<(TB * DE + 255) / 256, 256, 0, stream>>>(ef, ebs, rowmapb, emo);
    gemm(emo, 2 * DE, nullptr, lin_wh, 2 * DE, 0, lin_b, nullptr, 0, hid, DH,
         TB, DH, 2 * DE, INIT_BIAS, /*relu=*/1);
    fc_logsoftmax_kernel<<<(TB + 127) / 128, 128, 0, stream>>>(hid, fc_w, fc_b, (float*)d_out);
}
